// HeterogeneousHORNNetwork_29875792511035
// MI455X (gfx1250) — matrix operand order repaired and checked by reading, not compiled, not run on hardware
//
#include <hip/hip_runtime.h>
#include <hip/hip_bf16.h>

typedef __attribute__((ext_vector_type(16))) __bf16 v16bf;
typedef __attribute__((ext_vector_type(8)))  __bf16 v8bf;
typedef __attribute__((ext_vector_type(8)))  float  v8f;
typedef __attribute__((ext_vector_type(4)))  unsigned int v4u;
typedef __attribute__((ext_vector_type(8)))  int v8i;
typedef __attribute__((ext_vector_type(4)))  int v4i;

#define N_NODES  512
#define SEQ_T    1024
#define BATCH_B  256
#define OUT_C    10
#define K_CACHE  256   // K-range [0,256) of W_hh held in LDS (256 KB bf16)

__device__ __forceinline__ unsigned short f32_to_bf16(float f) {
    unsigned int u = __float_as_uint(f);
    u += 0x7FFFu + ((u >> 16) & 1u);       // round-to-nearest-even
    return (unsigned short)(u >> 16);
}

// Prep: convert W_hh (N x N, fp32 row-major) -> bf16 row-major in workspace.
__global__ void horn_prep_whh_bf16(const float* __restrict__ whh,
                                   unsigned short* __restrict__ out) {
    int i = blockIdx.x * blockDim.x + threadIdx.x;
    if (i < N_NODES * N_NODES) out[i] = f32_to_bf16(whh[i]);
}

// Persistent per-batch-tile recurrent kernel.
// grid = 16 workgroups (16 batch rows each), block = 256 threads = 8 waves.
// Wave w owns N-columns [64*w, 64*w+64) as 4 WMMA 16x16 tiles.
__global__ __launch_bounds__(256) void horn_kernel(
    const float* __restrict__ input,    // (B, T)
    const float* __restrict__ omega,    // (N)
    const float* __restrict__ gamma_,   // (N)
    const float* __restrict__ alpha,    // (N)
    const float* __restrict__ w_ih_w,   // (N, 1)
    const float* __restrict__ w_ih_b,   // (N)
    const float* __restrict__ w_hh_b,   // (N)
    const unsigned short* __restrict__ whh_bf, // (N, N) bf16 row-major
    const float* __restrict__ ro_w,     // (10, N)
    const float* __restrict__ ro_b,     // (10)
    float* __restrict__ out)            // (B, 10)
{
    // LDS: 256 KB W cache (K half, all n) + 16 KB y tile + stage = ~272 KB.
    __shared__ unsigned short wCache[N_NODES * K_CACHE]; // wCache[n*256 + k], k<256
    __shared__ unsigned short yLDS[16 * N_NODES];        // y tile in bf16
    __shared__ float          sStage[16];                // per-step inputs

    const int tid  = threadIdx.x;
    const int wave = tid >> 5;
    const int lane = tid & 31;
    const int nl   = lane & 15;
    const int half = lane >> 4;                    // 0: lanes 0-15, 1: lanes 16-31
    const int b0   = blockIdx.x * 16;
    const float inv_sqrt_n = 0.044194173824159216f;  // 1/sqrt(512)

    // ---- TDM: DMA the K-half of W_hh (512 rows x 512B from 1024B-stride
    //      tensor) into the LDS cache. One wave issues; all wait at barrier.
    if (wave == 0) {
        unsigned long long gaddr = (unsigned long long)(const void*)whh_bf;
        unsigned int ldsAddr = (unsigned int)(unsigned long long)(void*)wCache;
        v4u g0;
        g0.x = 1u;                                  // count=1 (valid), user mode
        g0.y = ldsAddr;                             // lds_addr (bits 63:32)
        g0.z = (unsigned int)gaddr;                 // global_addr[31:0]
        g0.w = (unsigned int)((gaddr >> 32) & 0x01FFFFFFu) | (2u << 30); // +type=2
        v8i g1;
        g1[0] = 0x00030000;                         // data_size = 3 (8-byte units)
        g1[1] = (int)(128u << 16);                  // tensor_dim0 = 128 units (1KB row)
        g1[2] = (int)(512u << 16);                  // tensor_dim1 = 512 rows
        g1[3] = (int)(64u  << 16);                  // tile_dim0 = 64 units (512B)
        g1[4] = 512;                                // tile_dim1 = 512 rows
        g1[5] = 128;                                // tensor_dim0_stride = 128 units
        g1[6] = 0;                                  // stride0 hi | stride1 lo
        g1[7] = 1;                                  // tensor_dim1_stride = 65536 units
        v4i gz4 = {};
        v8i gz8 = {};
        __builtin_amdgcn_tensor_load_to_lds(g0, g1, gz4, gz4, gz8, 0);
        __builtin_amdgcn_s_wait_tensorcnt(0);
    }

    // Per-lane constant node parameters for this lane's 4 n-columns.
    float k_win[4], k_bias[4], k_alpha[4], k_g2[4], k_om2[4];
    int   k_n[4];
    #pragma unroll
    for (int tt = 0; tt < 4; ++tt) {
        int n = wave * 64 + tt * 16 + nl;
        k_n[tt]     = n;
        k_win[tt]   = w_ih_w[n];
        k_bias[tt]  = w_ih_b[n] + w_hh_b[n];
        k_alpha[tt] = alpha[n];
        k_g2[tt]    = 2.0f * gamma_[n];
        float om    = omega[n];
        k_om2[tt]   = om * om;
    }

    // y0 = 0 in LDS (bf16), x/y register state = 0.
    for (int i = tid; i < 16 * N_NODES; i += 256) yLDS[i] = 0;

    float xr[4][8], yr[4][8];
    #pragma unroll
    for (int tt = 0; tt < 4; ++tt)
        #pragma unroll
        for (int r = 0; r < 8; ++r) { xr[tt][r] = 0.0f; yr[tt][r] = 0.0f; }

    // A-fragment addressing (16-bit A 16x32 layout, contiguous K order):
    // lanes 0-15 : M=lane,    K {0..15} of the 32-chunk
    // lanes 16-31: M=lane-16, K {16..31}
    const int m_a     = nl;
    const int run_off = half ? 16 : 0;
    const int koffB   = half ? 16 : 0;   // B-fragment K offset within 32-chunk

    for (int t = 0; t < SEQ_T; ++t) {
        if (tid < 16) sStage[tid] = input[(b0 + tid) * SEQ_T + t];
        __syncthreads();   // y(t-1), sStage (and on t==0 the TDM fill) visible

        v8f acc[4] = {};   // I_rec accumulators (f32), one per n-tile

        // Prefetch first streamed-K B fragments (K = 256..287) from L2.
        v16bf bnext[4];
        #pragma unroll
        for (int tt = 0; tt < 4; ++tt)
            bnext[tt] = *(const v16bf*)&whh_bf[k_n[tt] * N_NODES + K_CACHE + koffB];

        // ---- Phase 1: K in [0,256) with B from the LDS cache --------------
        for (int kc = 0; kc < 8; ++kc) {
            const int kbase = kc * 32;
            const v8bf alo = *(const v8bf*)&yLDS[m_a * N_NODES + kbase + run_off];
            const v8bf ahi = *(const v8bf*)&yLDS[m_a * N_NODES + kbase + run_off + 8];
            v16bf afrag;
            #pragma unroll
            for (int i = 0; i < 8; ++i) { afrag[i] = alo[i]; afrag[8 + i] = ahi[i]; }

            #pragma unroll
            for (int tt = 0; tt < 4; ++tt) {
                const int bi = k_n[tt] * K_CACHE + kbase + koffB;
                const v8bf blo = *(const v8bf*)&wCache[bi];
                const v8bf bhi = *(const v8bf*)&wCache[bi + 8];
                v16bf bfrag;
                #pragma unroll
                for (int i = 0; i < 8; ++i) { bfrag[i] = blo[i]; bfrag[8 + i] = bhi[i]; }
                acc[tt] = __builtin_amdgcn_wmma_f32_16x16x32_bf16(
                    false, afrag, false, bfrag, (short)0, acc[tt], false, false);
            }
        }

        // ---- Phase 2: K in [256,512) streamed from L2, double-buffered ----
        for (int kc = 8; kc < 16; ++kc) {
            const int kbase = kc * 32;
            v16bf bcur[4];
            #pragma unroll
            for (int tt = 0; tt < 4; ++tt) bcur[tt] = bnext[tt];
            if (kc < 15) {
                #pragma unroll
                for (int tt = 0; tt < 4; ++tt)
                    bnext[tt] = *(const v16bf*)
                        &whh_bf[k_n[tt] * N_NODES + kbase + 32 + koffB];
            }
            const v8bf alo = *(const v8bf*)&yLDS[m_a * N_NODES + kbase + run_off];
            const v8bf ahi = *(const v8bf*)&yLDS[m_a * N_NODES + kbase + run_off + 8];
            v16bf afrag;
            #pragma unroll
            for (int i = 0; i < 8; ++i) { afrag[i] = alo[i]; afrag[8 + i] = ahi[i]; }

            #pragma unroll
            for (int tt = 0; tt < 4; ++tt)
                acc[tt] = __builtin_amdgcn_wmma_f32_16x16x32_bf16(
                    false, afrag, false, bcur[tt], (short)0, acc[tt], false, false);
        }

        // Pull this lane's 8 batch inputs (m = r + 8*half) before the barrier.
        float sm[8];
        #pragma unroll
        for (int r = 0; r < 8; ++r) sm[r] = sStage[r + 8 * half];

        __syncthreads();   // all yLDS/sStage reads of step t complete

        // Elementwise oscillator update; C/D layout: lane holds (m = r+8*half, n).
        #pragma unroll
        for (int tt = 0; tt < 4; ++tt) {
            #pragma unroll
            for (int r = 0; r < 8; ++r) {
                float Irec  = acc[tt][r];
                float A     = (sm[r] * k_win[tt] + k_bias[tt] + Irec) * inv_sqrt_n;
                float e     = __expf(2.0f * A);
                float th    = 1.0f - 2.0f / (e + 1.0f);   // tanh, saturates cleanly
                float yo    = yr[tt][r], xo = xr[tt][r];
                float accel = k_alpha[tt] * th - k_g2[tt] * yo - k_om2[tt] * xo;
                float xn = xo + yo;
                float yn = yo + accel;
                xr[tt][r] = xn;
                yr[tt][r] = yn;
                int m = r + 8 * half;
                yLDS[m * N_NODES + k_n[tt]] = f32_to_bf16(yn);
            }
        }
    }

    // Final x dump reuses the (now dead) W cache as fp32 scratch.
    float* xLDSf = (float*)wCache;
    #pragma unroll
    for (int tt = 0; tt < 4; ++tt)
        #pragma unroll
        for (int r = 0; r < 8; ++r) {
            int m = r + 8 * half;
            xLDSf[m * N_NODES + k_n[tt]] = xr[tt][r];
        }
    __syncthreads();

    // Readout: 16 rows x 10 classes = 160 dot products of length 512.
    if (tid < 16 * OUT_C) {
        int m = tid / OUT_C, c = tid % OUT_C;
        float a = ro_b[c];
        for (int n = 0; n < N_NODES; ++n)
            a += xLDSf[m * N_NODES + n] * ro_w[c * N_NODES + n];
        out[(b0 + m) * OUT_C + c] = a;
    }
}

extern "C" void kernel_launch(void* const* d_in, const int* in_sizes, int n_in,
                              void* d_out, int out_size, void* d_ws, size_t ws_size,
                              hipStream_t stream) {
    const float* input   = (const float*)d_in[0];  // (256, 1024)
    const float* omega   = (const float*)d_in[1];  // (512)
    const float* gamma_  = (const float*)d_in[2];  // (512)
    const float* alpha   = (const float*)d_in[3];  // (512)
    const float* w_ih_w  = (const float*)d_in[4];  // (512, 1)
    const float* w_ih_b  = (const float*)d_in[5];  // (512)
    const float* w_hh_w  = (const float*)d_in[6];  // (512, 512)
    const float* w_hh_b  = (const float*)d_in[7];  // (512)
    const float* ro_w    = (const float*)d_in[8];  // (10, 512)
    const float* ro_b    = (const float*)d_in[9];  // (10)
    float* outp          = (float*)d_out;          // (256, 10)

    unsigned short* whh_bf = (unsigned short*)d_ws;  // 512 KB bf16 copy of W_hh

    horn_prep_whh_bf16<<<(N_NODES * N_NODES + 255) / 256, 256, 0, stream>>>(
        w_hh_w, whh_bf);

    horn_kernel<<<BATCH_B / 16, 256, 0, stream>>>(
        input, omega, gamma_, alpha, w_ih_w, w_ih_b, w_hh_b,
        whh_bf, ro_w, ro_b, outp);
}
